// AttnModel_23794118820305
// MI455X (gfx1250) — matrix operand order repaired and checked by reading, not compiled, not run on hardware
//
#include <hip/hip_runtime.h>
#include <hip/hip_bf16.h>

// ---------------------------------------------------------------------------
// Types for CDNA5 WMMA (gfx1250, wave32)
// ---------------------------------------------------------------------------
typedef __attribute__((ext_vector_type(16))) __bf16 v16bf;
typedef __attribute__((ext_vector_type(8)))  float  v8f;
typedef __attribute__((ext_vector_type(4)))  unsigned int u32x4;

struct B32x2 { u32x4 a, b; };

__device__ __forceinline__ __bf16 f2bf(float f) {
    unsigned u = __builtin_bit_cast(unsigned, f);
    unsigned short s = (unsigned short)((u + 0x7FFFu + ((u >> 16) & 1u)) >> 16);
    return __builtin_bit_cast(__bf16, s);
}
__device__ __forceinline__ float bf2f(__bf16 b) {
    unsigned short s = __builtin_bit_cast(unsigned short, b);
    return __builtin_bit_cast(float, (unsigned)s << 16);
}

// A fragment: 16x32 bf16, row-major source [lda].  ISA 7.12.2:
//  lanes 0-15 : M=lane,    K = {k0..k0+7} U {k0+16..k0+23}
//  lanes 16-31: M=lane-16, K = {k0+8..k0+15} U {k0+24..k0+31}
__device__ __forceinline__ v16bf load_a_frag(const __bf16* A, int lda, int m0,
                                             int k0, int lane) {
    int row = m0 + (lane & 15);
    int kk  = k0 + ((lane & 16) ? 8 : 0);
    const __bf16* p = A + (size_t)row * lda + kk;
    B32x2 t;
    t.a = *(const u32x4*)(p);       // 8 bf16: K kk..kk+7
    t.b = *(const u32x4*)(p + 16);  // 8 bf16: K kk+16..kk+23
    return __builtin_bit_cast(v16bf, t);
}

// B fragment: 32x16 bf16 (KxN), weight stored [N][K] row-major (= torch [out][in]):
//  lanes 0-15 : N=n0+lane,    K = {k0..k0+7} U {k0+16..k0+23}
//  lanes 16-31: N=n0+lane-16, K = {k0+8..k0+15} U {k0+24..k0+31}
__device__ __forceinline__ v16bf load_b_frag(const __bf16* W, int ldw, int n0,
                                             int k0, int lane) {
    int n  = n0 + (lane & 15);
    int kk = k0 + ((lane & 16) ? 8 : 0);
    const __bf16* p = W + (size_t)n * ldw + kk;
    B32x2 t;
    t.a = *(const u32x4*)(p);
    t.b = *(const u32x4*)(p + 16);
    return __builtin_bit_cast(v16bf, t);
}

#define WMMA_BF16(a, b, c) \
    __builtin_amdgcn_wmma_f32_16x16x32_bf16(false, (a), false, (b), (short)0, (c), false, false)

// ---------------------------------------------------------------------------
// Problem constants
// ---------------------------------------------------------------------------
#define BATCH 16
#define NS    32
#define NGH   2048
#define NBRS  64
#define DM    1024
#define NH    8
#define DK    128
#define ROWS  (BATCH * NS)        // 512 source rows
#define KROWS (BATCH * NGH)       // 32768 neighbor rows

// ---------------------------------------------------------------------------
// Elementwise prep kernels
// ---------------------------------------------------------------------------
__global__ void build_qcat_k(const float* __restrict__ src,
                             const float* __restrict__ src_t,
                             const float* __restrict__ src_p,
                             float* __restrict__ qf, __bf16* __restrict__ qb) {
    int n = ROWS * DM;
    for (int i = blockIdx.x * blockDim.x + threadIdx.x; i < n;
         i += gridDim.x * blockDim.x) {
        int r = i >> 10, c = i & 1023;
        float v;
        if (c < 256)       v = src[r * 256 + c];
        else if (c < 512)  v = 0.0f;                 // src_e = zeros
        else if (c < 768)  v = src_t[r * 256 + (c - 512)];
        else               v = src_p[r * 256 + (c - 768)];
        qf[i] = v;
        qb[i] = f2bf(v);
    }
}

__global__ void build_kcat_k(const float* __restrict__ seq,
                             const float* __restrict__ seq_e,
                             const float* __restrict__ seq_t,
                             const float* __restrict__ seq_p,
                             __bf16* __restrict__ kb) {
    int n = KROWS * DM;
    for (int i = blockIdx.x * blockDim.x + threadIdx.x; i < n;
         i += gridDim.x * blockDim.x) {
        int r = i >> 10, c = i & 1023;
        float v;
        if (c < 256)       v = seq[r * 256 + c];
        else if (c < 512)  v = seq_e[r * 256 + (c - 256)];
        else if (c < 768)  v = seq_t[r * 256 + (c - 512)];
        else               v = seq_p[r * 256 + (c - 768)];
        kb[i] = f2bf(v);
    }
}

__global__ void f32_to_bf16_k(const float* __restrict__ in,
                              __bf16* __restrict__ out, int n) {
    for (int i = blockIdx.x * blockDim.x + threadIdx.x; i < n;
         i += gridDim.x * blockDim.x)
        out[i] = f2bf(in[i]);
}

// ---------------------------------------------------------------------------
// Generic bf16 WMMA GEMM:  C[M,N] = A[M,K] * W[N,K]^T  (+bias +resid, relu)
// 128 threads = 4 waves arranged 2x2; each wave owns a 32x32 macro tile
// (2x2 WMMA tiles, A/B fragments reused twice).  Software pipeline with the
// final k-step peeled: loads for k0 are issued before the WMMAs for k0-32,
// no per-iteration branch, no fragment-copy tail.
// grid = (N/64, M/64).  M,N multiples of 64; K multiple of 32.
// ---------------------------------------------------------------------------
__global__ __launch_bounds__(128)
void wmma_gemm_k(const __bf16* __restrict__ A, const __bf16* __restrict__ W,
                 float* __restrict__ Cf, __bf16* __restrict__ Cbf,
                 const float* __restrict__ bias,
                 const float* __restrict__ resid,
                 int M, int N, int K, int lda, int relu) {
    int lane = threadIdx.x & 31;
    int wave = threadIdx.x >> 5;
    int m0 = blockIdx.y * 64 + (wave >> 1) * 32;
    int n0 = blockIdx.x * 64 + (wave & 1) * 32;

    v8f acc[2][2];
#pragma unroll
    for (int i = 0; i < 2; ++i)
#pragma unroll
        for (int j = 0; j < 2; ++j)
#pragma unroll
            for (int e = 0; e < 8; ++e) acc[i][j][e] = 0.0f;

    v16bf a0 = load_a_frag(A, lda, m0,      0, lane);
    v16bf a1 = load_a_frag(A, lda, m0 + 16, 0, lane);
    v16bf b0 = load_b_frag(W, K,   n0,      0, lane);
    v16bf b1 = load_b_frag(W, K,   n0 + 16, 0, lane);

    for (int k0 = 32; k0 < K; k0 += 32) {
        v16bf na0 = load_a_frag(A, lda, m0,      k0, lane);
        v16bf na1 = load_a_frag(A, lda, m0 + 16, k0, lane);
        v16bf nb0 = load_b_frag(W, K,   n0,      k0, lane);
        v16bf nb1 = load_b_frag(W, K,   n0 + 16, k0, lane);
        acc[0][0] = WMMA_BF16(a0, b0, acc[0][0]);
        acc[0][1] = WMMA_BF16(a0, b1, acc[0][1]);
        acc[1][0] = WMMA_BF16(a1, b0, acc[1][0]);
        acc[1][1] = WMMA_BF16(a1, b1, acc[1][1]);
        a0 = na0; a1 = na1; b0 = nb0; b1 = nb1;
    }
    acc[0][0] = WMMA_BF16(a0, b0, acc[0][0]);
    acc[0][1] = WMMA_BF16(a0, b1, acc[0][1]);
    acc[1][0] = WMMA_BF16(a1, b0, acc[1][0]);
    acc[1][1] = WMMA_BF16(a1, b1, acc[1][1]);

    // D layout: lanes 0-15 -> M=mt+r, lanes 16-31 -> M=mt+8+r; N = nt+(lane&15)
#pragma unroll
    for (int i = 0; i < 2; ++i) {
#pragma unroll
        for (int j = 0; j < 2; ++j) {
            int n  = n0 + j * 16 + (lane & 15);
            int mb = m0 + i * 16 + ((lane & 16) ? 8 : 0);
            float bv = bias ? bias[n] : 0.0f;
#pragma unroll
            for (int r = 0; r < 8; ++r) {
                size_t o = (size_t)(mb + r) * N + n;
                float v = acc[i][j][r] + bv;
                if (resid) v += resid[o];
                if (relu)  v = fmaxf(v, 0.0f);
                if (Cf)  Cf[o]  = v;
                if (Cbf) Cbf[o] = f2bf(v);
            }
        }
    }
}

// ---------------------------------------------------------------------------
// Fused per-(b,s,head) attention kernel.  grid = B*NS*H = 4096, block = 256.
// Each WG: K_h,V_h [64x128] = WMMA projections of the 64 private neighbors
// (software-pipelined: 1 A-frag + 8 B-frags in flight under 8 WMMAs), then
// scores -> mask -> softmax -> attn*V.  The 256-byte Q head vector is staged
// via the CDNA5 async global->LDS pipe (global_load_async_to_lds_b128,
// tracked by ASYNCcnt).
// ---------------------------------------------------------------------------
__global__ __launch_bounds__(256)
void attn_fused_k(const __bf16* __restrict__ kcat,
                  const __bf16* __restrict__ wk,
                  const __bf16* __restrict__ wv,
                  const __bf16* __restrict__ Qbf,
                  const unsigned char* __restrict__ mask,
                  __bf16* __restrict__ attn_out,   // [ROWS][DM] bf16
                  float* __restrict__ attn_probs)  // [ROWS][H][NBRS] f32
{
    __shared__ __bf16 Kh[NBRS][DK + 4];
    __shared__ __bf16 Vh[NBRS][DK + 4];
    __shared__ __bf16 Qh[DK];
    __shared__ float  sc[NBRS];
    __shared__ float  pr[NBRS];
    __shared__ float  red[2];

    int wg = blockIdx.x;
    int h  = wg & (NH - 1);
    int bs = wg >> 3;                 // b*NS + s
    int b  = bs >> 5;
    int s  = bs & 31;

    int tid  = threadIdx.x;
    int lane = tid & 31;
    int wave = tid >> 5;

    // async DMA of the Q head vector (16 lanes x 16B = 256B) into LDS
    if (tid < 16) {
        unsigned lds_off = (unsigned)(uintptr_t)(&Qh[tid * 8]);
        const __bf16* gp = Qbf + (size_t)bs * DM + h * DK + tid * 8;
        asm volatile("global_load_async_to_lds_b128 %0, %1, off"
                     :: "v"(lds_off), "v"(gp) : "memory");
    }

    // ---- projections: M=64 neighbors, N=128 head dims, K=1024 ----
    const __bf16* Ablk = kcat + ((size_t)b * NGH + (size_t)s * NBRS) * DM;
    int mt  = wave & 3;               // M-tile 0..3
    int ntb = (wave >> 2) * 4;        // N-tile base 0 or 4

    v8f accK[4], accV[4];
#pragma unroll
    for (int j = 0; j < 4; ++j)
#pragma unroll
        for (int i = 0; i < 8; ++i) { accK[j][i] = 0.0f; accV[j][i] = 0.0f; }

    v16bf a, bk[4], bv[4];
    a = load_a_frag(Ablk, DM, mt * 16, 0, lane);
#pragma unroll
    for (int j = 0; j < 4; ++j) {
        int ncol = h * DK + (ntb + j) * 16;
        bk[j] = load_b_frag(wk, DM, ncol, 0, lane);
        bv[j] = load_b_frag(wv, DM, ncol, 0, lane);
    }

    for (int k0 = 32; k0 < DM; k0 += 32) {
        v16bf na = load_a_frag(Ablk, DM, mt * 16, k0, lane);
        v16bf nbk[4], nbv[4];
#pragma unroll
        for (int j = 0; j < 4; ++j) {
            int ncol = h * DK + (ntb + j) * 16;
            nbk[j] = load_b_frag(wk, DM, ncol, k0, lane);
            nbv[j] = load_b_frag(wv, DM, ncol, k0, lane);
        }
#pragma unroll
        for (int j = 0; j < 4; ++j) {
            accK[j] = WMMA_BF16(a, bk[j], accK[j]);
            accV[j] = WMMA_BF16(a, bv[j], accV[j]);
        }
        a = na;
#pragma unroll
        for (int j = 0; j < 4; ++j) { bk[j] = nbk[j]; bv[j] = nbv[j]; }
    }
#pragma unroll
    for (int j = 0; j < 4; ++j) {
        accK[j] = WMMA_BF16(a, bk[j], accK[j]);
        accV[j] = WMMA_BF16(a, bv[j], accV[j]);
    }

#pragma unroll
    for (int j = 0; j < 4; ++j) {
        int col = (ntb + j) * 16 + (lane & 15);
        int rb  = mt * 16 + ((lane & 16) ? 8 : 0);
#pragma unroll
        for (int r = 0; r < 8; ++r) {
            Kh[rb + r][col] = f2bf(accK[j][r]);
            Vh[rb + r][col] = f2bf(accV[j][r]);
        }
    }
    asm volatile("s_wait_asynccnt 0x0" ::: "memory");   // Q DMA complete
    __syncthreads();

    // ---- scores: 4 threads per neighbor, 32 dims each, shfl reduce ----
    {
        int n  = tid >> 2;
        int qq = tid & 3;
        float part = 0.0f;
#pragma unroll
        for (int d = 0; d < 32; ++d)
            part += bf2f(Kh[n][qq * 32 + d]) * bf2f(Qh[qq * 32 + d]);
        part += __shfl_xor(part, 1, 32);
        part += __shfl_xor(part, 2, 32);
        if (qq == 0) sc[n] = part;
    }
    __syncthreads();

    // ---- scale, mask, softmax (over 64 neighbors) ----
    if (tid < NBRS) {
        float v = sc[tid] * 0.08838834764831845f;   // 1/sqrt(128)
        if (mask[(size_t)b * NGH + s * NBRS + tid]) v = -1e10f;
        sc[tid] = v;
    }
    __syncthreads();
    if (tid == 0) {
        float mx = -3.4e38f;
        for (int i = 0; i < NBRS; ++i) mx = fmaxf(mx, sc[i]);
        float sum = 0.0f;
        for (int i = 0; i < NBRS; ++i) sum += __expf(sc[i] - mx);
        red[0] = mx;
        red[1] = sum;
    }
    __syncthreads();
    if (tid < NBRS) {
        float p = __expf(sc[tid] - red[0]) / red[1];
        pr[tid] = p;
        attn_probs[((size_t)bs * NH + h) * NBRS + tid] = p;
    }
    __syncthreads();

    // ---- out = attn * V_h  (128 dims, one thread each) ----
    if (tid < DK) {
        float acc = 0.0f;
#pragma unroll 8
        for (int n = 0; n < NBRS; ++n) acc += pr[n] * bf2f(Vh[n][tid]);
        attn_out[(size_t)bs * DM + h * DK + tid] = f2bf(acc);
    }
}

// ---------------------------------------------------------------------------
// LayerNorm over DM + write bf16 concat row [normed(1024) | src(256)]
// ---------------------------------------------------------------------------
__global__ __launch_bounds__(256)
void ln_concat_k(const float* __restrict__ res1, const float* __restrict__ src,
                 const float* __restrict__ g, const float* __restrict__ bta,
                 __bf16* __restrict__ xcat) {
    __shared__ float sb[256];
    int row = blockIdx.x;
    int tid = threadIdx.x;

    float s = 0.0f, ss = 0.0f;
    for (int c = tid; c < DM; c += 256) {
        float v = res1[(size_t)row * DM + c];
        s += v; ss += v * v;
    }
    sb[tid] = s; __syncthreads();
    for (int o = 128; o > 0; o >>= 1) { if (tid < o) sb[tid] += sb[tid + o]; __syncthreads(); }
    float mean = sb[0] * (1.0f / DM);
    __syncthreads();
    sb[tid] = ss; __syncthreads();
    for (int o = 128; o > 0; o >>= 1) { if (tid < o) sb[tid] += sb[tid + o]; __syncthreads(); }
    float var  = sb[0] * (1.0f / DM) - mean * mean;
    float rstd = rsqrtf(var + 1e-5f);

    for (int c = tid; c < DM; c += 256) {
        float v = (res1[(size_t)row * DM + c] - mean) * rstd * g[c] + bta[c];
        xcat[(size_t)row * (DM + 256) + c] = f2bf(v);
    }
    xcat[(size_t)row * (DM + 256) + DM + tid] = f2bf(src[(size_t)row * 256 + tid]);
}

// ---------------------------------------------------------------------------
// Host-side orchestration
// ---------------------------------------------------------------------------
extern "C" void kernel_launch(void* const* d_in, const int* in_sizes, int n_in,
                              void* d_out, int out_size, void* d_ws, size_t ws_size,
                              hipStream_t stream) {
    const float* src   = (const float*)d_in[0];
    const float* src_t = (const float*)d_in[1];
    const float* src_p = (const float*)d_in[2];
    const float* seq   = (const float*)d_in[3];
    const float* seq_t = (const float*)d_in[4];
    const float* seq_e = (const float*)d_in[5];
    const float* seq_p = (const float*)d_in[6];
    const unsigned char* mask = (const unsigned char*)d_in[7];
    const float* w_qs  = (const float*)d_in[8];
    const float* w_ks  = (const float*)d_in[9];
    const float* w_vs  = (const float*)d_in[10];
    const float* fc_w  = (const float*)d_in[11];
    const float* fc_b  = (const float*)d_in[12];
    const float* ln_g  = (const float*)d_in[13];
    const float* ln_b  = (const float*)d_in[14];
    const float* fc1_w = (const float*)d_in[15];
    const float* fc1_b = (const float*)d_in[16];
    const float* fc2_w = (const float*)d_in[17];
    const float* fc2_b = (const float*)d_in[18];

    char* ws = (char*)d_ws;
    size_t off = 0;
    auto alloc = [&](size_t bytes) -> void* {
        void* p = ws + off;
        off = (off + bytes + 255) & ~(size_t)255;
        return p;
    };

    float*  qcat    = (float*) alloc((size_t)ROWS * DM * 4);          // f32 q (residual)
    __bf16* qcat_bf = (__bf16*)alloc((size_t)ROWS * DM * 2);
    __bf16* kcat_bf = (__bf16*)alloc((size_t)KROWS * DM * 2);         // 64 MiB
    __bf16* wq_bf   = (__bf16*)alloc((size_t)DM * DM * 2);
    __bf16* wk_bf   = (__bf16*)alloc((size_t)DM * DM * 2);
    __bf16* wv_bf   = (__bf16*)alloc((size_t)DM * DM * 2);
    __bf16* fcw_bf  = (__bf16*)alloc((size_t)DM * DM * 2);
    __bf16* fc1w_bf = (__bf16*)alloc((size_t)256 * (DM + 256) * 2);
    __bf16* fc2w_bf = (__bf16*)alloc((size_t)256 * 256 * 2);
    __bf16* Qbf     = (__bf16*)alloc((size_t)ROWS * DM * 2);
    __bf16* ao_bf   = (__bf16*)alloc((size_t)ROWS * DM * 2);          // attention out
    float*  res1    = (float*) alloc((size_t)ROWS * DM * 4);          // fc out + resid
    __bf16* xcat_bf = (__bf16*)alloc((size_t)ROWS * (DM + 256) * 2);
    __bf16* h_bf    = (__bf16*)alloc((size_t)ROWS * 256 * 2);

    float* z_out    = (float*)d_out;                                  // [512][256]
    float* attn_out = (float*)d_out + (size_t)ROWS * 256;             // [512][8][64]

    // 1. prep / precision conversion
    build_qcat_k<<<1024, 256, 0, stream>>>(src, src_t, src_p, qcat, qcat_bf);
    build_kcat_k<<<8192, 256, 0, stream>>>(seq, seq_e, seq_t, seq_p, kcat_bf);
    f32_to_bf16_k<<<2048, 256, 0, stream>>>(w_qs,  wq_bf,   DM * DM);
    f32_to_bf16_k<<<2048, 256, 0, stream>>>(w_ks,  wk_bf,   DM * DM);
    f32_to_bf16_k<<<2048, 256, 0, stream>>>(w_vs,  wv_bf,   DM * DM);
    f32_to_bf16_k<<<2048, 256, 0, stream>>>(fc_w,  fcw_bf,  DM * DM);
    f32_to_bf16_k<<<1024, 256, 0, stream>>>(fc1_w, fc1w_bf, 256 * (DM + 256));
    f32_to_bf16_k<<<256,  256, 0, stream>>>(fc2_w, fc2w_bf, 256 * 256);

    // 2. Q projection: [512,1024] x [1024,1024]^T -> bf16
    wmma_gemm_k<<<dim3(DM / 64, ROWS / 64), 128, 0, stream>>>(
        qcat_bf, wq_bf, nullptr, Qbf, nullptr, nullptr, ROWS, DM, DM, DM, 0);

    // 3. fused K/V projection + attention per (b,s,head)
    attn_fused_k<<<ROWS * NH, 256, 0, stream>>>(
        kcat_bf, wk_bf, wv_bf, Qbf, mask, ao_bf, attn_out);

    // 4. output fc + bias + residual(q) -> res1 (f32)
    wmma_gemm_k<<<dim3(DM / 64, ROWS / 64), 128, 0, stream>>>(
        ao_bf, fcw_bf, res1, nullptr, fc_b, qcat, ROWS, DM, DM, DM, 0);

    // 5. LayerNorm + concat with src -> xcat (bf16 [512][1280])
    ln_concat_k<<<ROWS, 256, 0, stream>>>(res1, src, ln_g, ln_b, xcat_bf);

    // 6. merge fc1 (relu) : [512,1280] x [256,1280]^T -> bf16
    wmma_gemm_k<<<dim3(256 / 64, ROWS / 64), 128, 0, stream>>>(
        xcat_bf, fc1w_bf, nullptr, h_bf, fc1_b, nullptr, ROWS, 256, DM + 256,
        DM + 256, 1);

    // 7. merge fc2 : [512,256] x [256,256]^T -> z (f32, d_out)
    wmma_gemm_k<<<dim3(256 / 64, ROWS / 64), 128, 0, stream>>>(
        h_bf, fc2w_bf, z_out, nullptr, fc2_b, nullptr, ROWS, 256, 256, 256, 0);

    (void)in_sizes; (void)n_in; (void)out_size; (void)ws_size;
}
